// VectorQuantizer_34265249087766
// MI455X (gfx1250) — matrix operand order repaired and checked by reading, not compiled, not run on hardware
//
#include <hip/hip_runtime.h>

#define N_ROWS 16384
#define DIM 512
#define K_CODES 4096
#define ND (N_ROWS * DIM)
#define NTILES (K_CODES / 16)
#define HTILES (NTILES / 2)          // tiles per K-half (per block)

typedef __attribute__((ext_vector_type(16))) __bf16 v16bf;
typedef __attribute__((ext_vector_type(8)))  float  v8f;

union FragBF { v16bf v; uint4 q[2]; };

__device__ __forceinline__ unsigned short f2bf_rne(float f) {
  unsigned int u = __float_as_uint(f);
  u += 0x7FFFu + ((u >> 16) & 1u);   // round-to-nearest-even
  return (unsigned short)(u >> 16);
}

// Low 32 bits of a generic pointer to __shared__ == LDS byte offset
// (shared aperture base occupies bits [63:32]).
__device__ __forceinline__ unsigned lds_off_u32(const void* p) {
  return (unsigned)(unsigned long long)reinterpret_cast<uintptr_t>(p);
}

// gfx1250 async memory->LDS (VGLOBAL async encoding, GVS mode: saddr base + u32 voffset).
__device__ __forceinline__ void async_ld_b128(unsigned lds, unsigned off, unsigned long long sbase) {
  asm volatile("global_load_async_to_lds_b128 %0, %1, %2"
               :: "v"(lds), "v"(off), "s"(sbase) : "memory");
}
__device__ __forceinline__ void async_ld_b32(unsigned lds, unsigned off, unsigned long long sbase) {
  asm volatile("global_load_async_to_lds_b32 %0, %1, %2"
               :: "v"(lds), "v"(off), "s"(sbase) : "memory");
}
__device__ __forceinline__ void wait_async0() {
  asm volatile("s_wait_asynccnt 0x0" ::: "memory");
}

// ---------------------------------------------------------------- P1: inputs f32 -> bf16
__global__ void vq_cvt_inputs(const float* __restrict__ x, unsigned short* __restrict__ xbf) {
  const float4* x4 = (const float4*)x;
  ushort4* o4 = (ushort4*)xbf;
  for (int t = blockIdx.x * blockDim.x + threadIdx.x; t < ND / 4; t += gridDim.x * blockDim.x) {
    float4 v = x4[t];
    ushort4 o;
    o.x = f2bf_rne(v.x); o.y = f2bf_rne(v.y);
    o.z = f2bf_rne(v.z); o.w = f2bf_rne(v.w);
    o4[t] = o;
  }
}

// ---------------------------------------------------------------- P2: codes f32 -> bf16, + ||e||^2
__global__ void vq_prep_codes(const float* __restrict__ e, unsigned short* __restrict__ ebf,
                              float* __restrict__ enorm) {
  __shared__ float red[128];
  const int k = blockIdx.x;
  const float* row = e + (size_t)k * DIM;
  unsigned short* orow = ebf + (size_t)k * DIM;
  float s = 0.f;
  for (int d = threadIdx.x; d < DIM; d += 128) {
    float v = row[d];
    orow[d] = f2bf_rne(v);
    s += v * v;
  }
  red[threadIdx.x] = s;
  __syncthreads();
  for (int st = 64; st > 0; st >>= 1) {
    if (threadIdx.x < st) red[threadIdx.x] += red[threadIdx.x + st];
    __syncthreads();
  }
  if (threadIdx.x == 0) enorm[k] = red[0];
}

// ---------------------------------------------------------------- G: WMMA distance + argmin
// Grid = 64 row-groups x 2 codebook halves. Block: 256 threads = 8 waves; each wave owns
// TWO 16-row strips (two register-resident B operands, 256 VGPRs) so each LDS A-fragment
// feeds two independent WMMA chains (1:1 ds_load:wmma, halved LDS traffic, latency hiding).
// Code tiles are double-buffered in LDS via GLOBAL_LOAD_ASYNC_TO_LDS (ASYNCcnt-tracked).
__global__ void __launch_bounds__(256, 1) vq_argmin(
    const unsigned short* __restrict__ xbf,
    const unsigned short* __restrict__ ebf,
    const float* __restrict__ enorm,
    float* __restrict__ candBest,    // [2][N_ROWS]
    int* __restrict__ candIdx)       // [2][N_ROWS]
{
  __shared__ __align__(16) unsigned short sE[2][16 * DIM];  // 2 x 16KB code tiles
  __shared__ __align__(16) float sEn[2][16];

  const int tid  = threadIdx.x;
  const int wave = tid >> 5;
  const int lane = tid & 31;
  const int half = lane >> 4;   // lane group 0: lanes 0-15, 1: lanes 16-31
  const int lmod = lane & 15;

  const int rg  = blockIdx.x >> 1;       // row group (256 rows)
  const int kh  = blockIdx.x & 1;        // codebook half
  const int at0 = kh * HTILES;           // first absolute tile index
  const int r0a = rg * 256 + wave * 16;          // strip A base row
  const int r0b = rg * 256 + 128 + wave * 16;    // strip B base row

  const unsigned long long ebf_a = (unsigned long long)(uintptr_t)ebf;
  const unsigned long long en_a  = (unsigned long long)(uintptr_t)enorm;
  const unsigned sE_lds  = lds_off_u32(&sE[0][0]);
  const unsigned sEn_lds = lds_off_u32(&sEn[0][0]);

  // Issue one tile's async loads: 16KB as 4 x b128 per thread (+ 16 norm dwords).
  auto issue_tile = [&](int at, int buf) {
    const unsigned goff = (unsigned)at * (16 * DIM * 2) + (unsigned)tid * 16;
    const unsigned lds  = sE_lds + (unsigned)buf * (16 * DIM * 2) + (unsigned)tid * 16;
    #pragma unroll
    for (int i = 0; i < 4; ++i)
      async_ld_b128(lds + i * 4096, goff + i * 4096, ebf_a);
    if (tid < 16)
      async_ld_b32(sEn_lds + (unsigned)buf * 64 + (unsigned)tid * 4,
                   (unsigned)at * 64 + (unsigned)tid * 4, en_a);
  };

  // B fragments: two 16-row strips x full K=512 (bf16), resident in 2x128 VGPRs.
  // B layout (16-bit, 32x16): lane = N column, contiguous 16 K-values per lane-half.
  FragBF bfA[16], bfB[16];
  {
    const unsigned short* xra = xbf + (size_t)(r0a + lmod) * DIM + half * 16;
    const unsigned short* xrb = xbf + (size_t)(r0b + lmod) * DIM + half * 16;
    #pragma unroll
    for (int s = 0; s < 16; ++s) {
      const uint4* pa = (const uint4*)(xra + s * 32);
      const uint4* pb = (const uint4*)(xrb + s * 32);
      bfA[s].q[0] = pa[0]; bfA[s].q[1] = pa[1];
      bfB[s].q[0] = pb[0]; bfB[s].q[1] = pb[1];
    }
  }

  issue_tile(at0, 0);

  float bestA = 3.4e38f, bestB = 3.4e38f;
  int   bidxA = 0,       bidxB = 0;

  for (int tt = 0; tt < HTILES; ++tt) {
    const int cur = tt & 1;
    wait_async0();        // this wave's tile data is in LDS
    __syncthreads();      // (a) everyone's tile landed  (b) buffer cur^1 free to overwrite
    if (tt + 1 < HTILES) issue_tile(at0 + tt + 1, cur ^ 1);

    // A layout (16-bit, 16x32): lane row m = lmod; elems 0..15 at K = s*32 + half*16 + e
    // (contiguous, matching B). Each A fragment feeds both strips.
    const unsigned short* abase = &sE[cur][lmod * DIM + half * 16];
    v8f accA = {}, accB = {};
    #pragma unroll
    for (int s = 0; s < 16; ++s) {
      FragBF a;
      const unsigned short* ap = abase + s * 32;
      a.q[0] = *(const uint4*)(ap);
      a.q[1] = *(const uint4*)(ap + 8);
      accA = __builtin_amdgcn_wmma_f32_16x16x32_bf16(
          false, a.v, false, bfA[s].v, (short)0, accA, false, false);
      accB = __builtin_amdgcn_wmma_f32_16x16x32_bf16(
          false, a.v, false, bfB[s].v, (short)0, accB, false, false);
    }

    // C layout: lane holds input row n = lmod; vgpr j holds code m = half*8 + j.
    // dist = ||e||^2 - 2*dot ; strict < keeps first (ties -> lowest index) within lane.
    const float4* en4 = (const float4*)&sEn[cur][half * 8];
    float4 e0 = en4[0], e1 = en4[1];
    float en[8] = { e0.x, e0.y, e0.z, e0.w, e1.x, e1.y, e1.z, e1.w };
    const int cbase = (at0 + tt) * 16 + half * 8;
    #pragma unroll
    for (int j = 0; j < 8; ++j) {
      float dA = fmaf(-2.0f, accA[j], en[j]);
      float dB = fmaf(-2.0f, accB[j], en[j]);
      if (dA < bestA) { bestA = dA; bidxA = cbase + j; }
      if (dB < bestB) { bestB = dB; bidxB = cbase + j; }
    }
    // no trailing barrier: buffer cur is only overwritten by issue(tt+2), reached only
    // after the tt+1 barrier (i.e. after this compute finished in every wave).
  }

  // Combine the two lane-halves holding the same input row (tie -> lower index).
  {
    float ob = __shfl_xor(bestA, 16, 32);
    int   oi = __shfl_xor(bidxA, 16, 32);
    if (ob < bestA || (ob == bestA && oi < bidxA)) { bestA = ob; bidxA = oi; }
    ob = __shfl_xor(bestB, 16, 32);
    oi = __shfl_xor(bidxB, 16, 32);
    if (ob < bestB || (ob == bestB && oi < bidxB)) { bestB = ob; bidxB = oi; }
  }

  if (half == 0) {
    candBest[(size_t)kh * N_ROWS + (r0a + lmod)] = bestA;
    candIdx [(size_t)kh * N_ROWS + (r0a + lmod)] = bidxA;
    candBest[(size_t)kh * N_ROWS + (r0b + lmod)] = bestB;
    candIdx [(size_t)kh * N_ROWS + (r0b + lmod)] = bidxB;
  }
}

// ---------------------------------------------------------------- Q: combine halves, gather, loss partials
__global__ void vq_gather(const float* __restrict__ x, const float* __restrict__ emb,
                          const float* __restrict__ candBest, const int* __restrict__ candIdx,
                          float* __restrict__ out, float* __restrict__ partial)
{
  __shared__ float red[256];
  const int tid = threadIdx.x;
  const int sub = tid >> 7;       // two rows processed in parallel
  const int l   = tid & 127;      // float4 lane within a 512-float row
  const int rbase = blockIdx.x * 64;
  const float4* x4 = (const float4*)x;
  const float4* e4 = (const float4*)emb;
  float4* o4 = (float4*)out;
  float acc = 0.f;
  for (int rr = sub; rr < 64; rr += 2) {
    const int row = rbase + rr;
    // combine the two codebook-half candidates (tie -> half 0 == lower index == first min)
    const float b0 = candBest[row], b1 = candBest[N_ROWS + row];
    const int   i0 = candIdx[row],  i1 = candIdx[N_ROWS + row];
    const int k = (b1 < b0) ? i1 : i0;
    if (l == 0) out[(size_t)ND + 1 + row] = (float)k;      // indices output
    float4 ev = e4[(size_t)k * 128 + l];
    float4 xv = x4[(size_t)row * 128 + l];
    o4[(size_t)row * 128 + l] = ev;             // straight-through output == quantized
    float dx = ev.x - xv.x, dy = ev.y - xv.y, dz = ev.z - xv.z, dw = ev.w - xv.w;
    acc += dx * dx + dy * dy + dz * dz + dw * dw;
  }
  red[tid] = acc;
  __syncthreads();
  for (int st = 128; st > 0; st >>= 1) {
    if (tid < st) red[tid] += red[tid + st];
    __syncthreads();
  }
  if (tid == 0) partial[blockIdx.x] = red[0];
}

// ---------------------------------------------------------------- F: final deterministic loss
__global__ void vq_loss(const float* __restrict__ partial, float* __restrict__ out) {
  __shared__ float red[256];
  const int tid = threadIdx.x;
  red[tid] = partial[tid];
  __syncthreads();
  for (int st = 128; st > 0; st >>= 1) {
    if (tid < st) red[tid] += red[tid + st];
    __syncthreads();
  }
  if (tid == 0) out[ND] = 1.25f * red[0] / (float)ND;  // (1 + 0.25) * mean((q-x)^2)
}

// ----------------------------------------------------------------
extern "C" void kernel_launch(void* const* d_in, const int* in_sizes, int n_in,
                              void* d_out, int out_size, void* d_ws, size_t ws_size,
                              hipStream_t stream) {
  const float* x   = (const float*)d_in[0];   // [8,2048,512] f32
  const float* emb = (const float*)d_in[1];   // [4096,512] f32
  float* out = (float*)d_out;                 // [quantized ND | loss 1 | indices N]

  char* ws = (char*)d_ws;
  unsigned short* xbf   = (unsigned short*)ws; ws += (size_t)ND * 2;            // 16 MB
  unsigned short* ebf   = (unsigned short*)ws; ws += (size_t)K_CODES * DIM * 2; // 4 MB
  float* enorm          = (float*)ws;          ws += (size_t)K_CODES * 4;
  float* candBest       = (float*)ws;          ws += (size_t)2 * N_ROWS * 4;
  int*   candIdx        = (int*)ws;            ws += (size_t)2 * N_ROWS * 4;
  float* partial        = (float*)ws;

  vq_cvt_inputs<<<2048, 256, 0, stream>>>(x, xbf);
  vq_prep_codes<<<K_CODES, 128, 0, stream>>>(emb, ebf, enorm);
  vq_argmin<<<128, 256, 0, stream>>>(xbf, ebf, enorm, candBest, candIdx);
  vq_gather<<<256, 256, 0, stream>>>(x, emb, candBest, candIdx, out, partial);
  vq_loss<<<1, 256, 0, stream>>>(partial, out);
}
